// ConvPointnet_42425686950060
// MI455X (gfx1250) — matrix operand order repaired and checked by reading, not compiled, not run on hardware
//
#include <hip/hip_runtime.h>

typedef _Float16 v16h __attribute__((ext_vector_type(16)));
typedef _Float16 v8h  __attribute__((ext_vector_type(8)));
typedef float    v8f  __attribute__((ext_vector_type(8)));

#define BB   4
#define TT   50000
#define NPTS (BB * TT)        // 200000
#define GRES 64
#define R2   4096
#define MT   64               // points per block tile
#define NBLK (NPTS / MT)      // 3125 (exact)

// ---------- monotone float<->uint encoding for atomicMax-based float max ----------
__device__ __forceinline__ unsigned enc_f32(float f) {
  unsigned u = __float_as_uint(f);
  return (u & 0x80000000u) ? ~u : (u | 0x80000000u);
}
__device__ __forceinline__ float dec_f32(unsigned k) {
  unsigned u = (k & 0x80000000u) ? (k & 0x7FFFFFFFu) : ~k;
  return __uint_as_float(u);
}

// single-instruction f16 relu (avoid select->maxnum canonicalize double-op)
__device__ __forceinline__ _Float16 relu_h(_Float16 v) {
#if __has_builtin(__builtin_fmaxf16)
  return __builtin_fmaxf16(v, (_Float16)0);
#else
  return v > (_Float16)0 ? v : (_Float16)0;
#endif
}
__device__ __forceinline__ float relu_f(float v) {
#if __has_builtin(__builtin_fmaxf)
  return __builtin_fmaxf(v, 0.f);
#else
  return v > 0.f ? v : 0.f;
#endif
}

// ---------- WMMA fragment helpers (CDNA5 16x16x32 f16) ----------
// A fragment: 16x32 tile from row-major [rows][ld] f16 buffer at (m0,k0).
// Per ISA 7.12.2: lane&15 = M row; lane>>4 selects K sub-chunk of 8.
__device__ __forceinline__ v16h load_a(const _Float16* x, int ld, int m0, int k0,
                                       int lane) {
  int m  = m0 + (lane & 15);
  int kh = (lane >> 4) * 8;
  const _Float16* p0 = x + (size_t)m * ld + k0 + kh;
  v8h lo = *(const v8h*)(p0);
  v8h hi = *(const v8h*)(p0 + 16);
  v16h a;
#pragma unroll
  for (int j = 0; j < 8; ++j) { a[j] = lo[j]; a[8 + j] = hi[j]; }
  return a;
}

// B fragment: 32x16 tile from row-major [K][ldn] f16 weights at (k0,n0); lane&15 = N col, lane>>4 selects K sub-chunk of 8.
__device__ __forceinline__ v16h load_b(const _Float16* w, int ldn, int k0, int n0,
                                       int lane) {
  int kh = (lane >> 4) * 8;
  const _Float16* p0 = w + (size_t)(k0 + kh) * ldn + n0 + (lane & 15);
  v16h b;
#pragma unroll
  for (int j = 0; j < 8; ++j) { b[j] = p0[(size_t)j * ldn]; b[8 + j] = p0[(size_t)(16 + j) * ldn]; }
  return b;
}

__device__ __forceinline__ v8f wmma16(v16h a, v16h b, v8f c) {
  return __builtin_amdgcn_wmma_f32_16x16x32_f16(false, a, false, b, (short)0, c,
                                                false, false);
}

// ---------- one resnet block on a 64-point tile staged in LDS ----------
// xs: 64x256 f16 (input, already staged+synced; DESTROYED: relu'd in place).
// ns: 64x128 f16 scratch. 8 waves; wave w owns output columns [16w, 16w+16).
__device__ __forceinline__ void resnet_tile(_Float16* xs, _Float16* ns,
                                            const _Float16* w0, const _Float16* w1,
                                            const _Float16* wsc,
                                            const float* b0, const float* b1,
                                            _Float16* netOut, int gp0) {
  const int lane  = threadIdx.x & 31;
  const int wv    = threadIdx.x >> 5;
  const int n0    = wv * 16;
  const int col   = lane & 15;
  const int rbase = (lane >> 4) * 8;

  // Phase A: shortcut GEMM on RAW x:  accs[ms] = xs @ wsc   (K=256)
  v8f accs[4];
  {
    v16h bs[8];
#pragma unroll
    for (int k = 0; k < 8; ++k) bs[k] = load_b(wsc, 128, 32 * k, n0, lane);
#pragma unroll
    for (int ms = 0; ms < 4; ++ms) {
      v8f a = {};
#pragma unroll
      for (int k = 0; k < 8; ++k)
        a = wmma16(load_a(xs, 256, 16 * ms, 32 * k, lane), bs[k], a);
      accs[ms] = a;
    }
  }
  __syncthreads();  // all waves done reading raw xs

  // Phase B: xs = relu(xs), one packed pass over LDS
  for (int e8 = threadIdx.x; e8 < MT * 256 / 8; e8 += 256) {
    v8h v = *(v8h*)&xs[(size_t)e8 * 8];
#pragma unroll
    for (int j = 0; j < 8; ++j) v[j] = relu_h(v[j]);
    *(v8h*)&xs[(size_t)e8 * 8] = v;
  }
  __syncthreads();

  // Phase C: ns = relu( relu_x @ w0 + b0 )   (K=256)
  {
    v16h bf[8];
#pragma unroll
    for (int k = 0; k < 8; ++k) bf[k] = load_b(w0, 128, 32 * k, n0, lane);
    float bias0 = b0[n0 + col];
#pragma unroll
    for (int ms = 0; ms < 4; ++ms) {
      v8f acc = {};
#pragma unroll
      for (int k = 0; k < 8; ++k)
        acc = wmma16(load_a(xs, 256, 16 * ms, 32 * k, lane), bf[k], acc);
#pragma unroll
      for (int r = 0; r < 8; ++r) {
        float v = relu_f(acc[r] + bias0);
        ns[(size_t)(16 * ms + rbase + r) * 128 + n0 + col] = (_Float16)v;
      }
    }
  }
  __syncthreads();

  // Phase D: out = accs + ns @ w1 + b1   (K=128)
  {
    v16h bw1[4];
#pragma unroll
    for (int k = 0; k < 4; ++k) bw1[k] = load_b(w1, 128, 32 * k, n0, lane);
    float bias1 = b1[n0 + col];
#pragma unroll
    for (int ms = 0; ms < 4; ++ms) {
      v8f a = accs[ms];
#pragma unroll
      for (int k = 0; k < 4; ++k)
        a = wmma16(load_a(ns, 128, 16 * ms, 32 * k, lane), bw1[k], a);
#pragma unroll
      for (int r = 0; r < 8; ++r) {
        int row = 16 * ms + rbase + r;
        netOut[(size_t)(gp0 + row) * 128 + n0 + col] = (_Float16)(a[r] + bias1);
      }
    }
  }
}

// ---------- kernels ----------
__global__ void k_stage_w(const float* __restrict__ f0, const float* __restrict__ f1,
                          const float* __restrict__ fs, const float* __restrict__ fc,
                          _Float16* __restrict__ w0h, _Float16* __restrict__ w1h,
                          _Float16* __restrict__ wsh, _Float16* __restrict__ wch) {
  int t = blockIdx.x * blockDim.x + threadIdx.x;
  const int N0 = 5 * 256 * 128, N1 = 5 * 128 * 128, NS = 5 * 256 * 128, NC = 128 * 512;
  if (t < N0) w0h[t] = (_Float16)f0[t];
  if (t < N1) w1h[t] = (_Float16)f1[t];
  if (t < NS) wsh[t] = (_Float16)fs[t];
  if (t < NC) wch[t] = (_Float16)fc[t];
}

__global__ void k_block0(const float* __restrict__ p, const float* __restrict__ wpos,
                         const float* __restrict__ bpos,
                         const _Float16* __restrict__ w0, const _Float16* __restrict__ w1,
                         const _Float16* __restrict__ wsc,
                         const float* __restrict__ b0, const float* __restrict__ b1,
                         int* __restrict__ idx, _Float16* __restrict__ netOut) {
  __shared__ __align__(16) _Float16 xs[MT * 256];
  __shared__ __align__(16) _Float16 ns[MT * 128];
  __shared__ float wS[3 * 256];
  __shared__ float bS[256];
  __shared__ float pS[MT * 3];

  const int tid = threadIdx.x;
  const int gp0 = blockIdx.x * MT;

  for (int i = tid; i < 3 * 256; i += 256) wS[i] = wpos[i];
  if (tid < 256) bS[tid] = bpos[tid];
  for (int i = tid; i < MT * 3; i += 256) pS[i] = p[(size_t)gp0 * 3 + i];
  __syncthreads();

  if (tid < MT) {
    float px = pS[tid * 3 + 0], py = pS[tid * 3 + 1], pz = pS[tid * 3 + 2];
    const float inv = 1.0f / (1.0f + 0.1f + 1e-5f);
    auto cell = [&](float a, float b2) {
      float xa = a * inv;
      xa = xa < 0.f ? 0.f : (xa > (1.f - 1e-5f) ? (1.f - 1e-5f) : xa);
      float xb = b2 * inv;
      xb = xb < 0.f ? 0.f : (xb > (1.f - 1e-5f) ? (1.f - 1e-5f) : xb);
      return (int)(xa * GRES) + GRES * (int)(xb * GRES);
    };
    int gp = gp0 + tid;
    idx[0 * NPTS + gp] = cell(px, pz);  // xz
    idx[1 * NPTS + gp] = cell(px, py);  // xy
    idx[2 * NPTS + gp] = cell(py, pz);  // yz
  }

  // h = p @ fc_pos_w + fc_pos_b  ->  xs (f16)
  for (int e = tid; e < MT * 256; e += 256) {
    int pt = e >> 8, c = e & 255;
    float h = bS[c] + pS[pt * 3 + 0] * wS[c] + pS[pt * 3 + 1] * wS[256 + c] +
              pS[pt * 3 + 2] * wS[512 + c];
    xs[e] = (_Float16)h;
  }
  __syncthreads();

  resnet_tile(xs, ns, w0, w1, wsc, b0, b1, netOut, gp0);
}

__global__ void k_scatter(const _Float16* __restrict__ net, const int* __restrict__ idx,
                          unsigned* __restrict__ grid) {
  int gp = blockIdx.x * blockDim.x + threadIdx.x;
  if (gp >= NPTS) return;
  int b = gp / TT;
  size_t base[3];
#pragma unroll
  for (int pl = 0; pl < 3; ++pl)
    base[pl] = ((size_t)(pl * BB + b) * R2 + idx[pl * NPTS + gp]) * 128;
  const _Float16* f = net + (size_t)gp * 128;
  for (int j = 0; j < 128; ++j) {
    unsigned key = enc_f32((float)f[j]);
#pragma unroll
    for (int pl = 0; pl < 3; ++pl) atomicMax(&grid[base[pl] + j], key);
  }
}

__global__ void k_block(const _Float16* __restrict__ netIn,
                        const unsigned* __restrict__ grid, const int* __restrict__ idx,
                        const _Float16* __restrict__ w0, const _Float16* __restrict__ w1,
                        const _Float16* __restrict__ wsc,
                        const float* __restrict__ b0, const float* __restrict__ b1,
                        _Float16* __restrict__ netOut) {
  __shared__ __align__(16) _Float16 xs[MT * 256];
  __shared__ __align__(16) _Float16 ns[MT * 128];
  __shared__ int cellS[3][MT];
  __shared__ int bS[MT];
  const int tid = threadIdx.x;
  const int gp0 = blockIdx.x * MT;

  if (tid < MT) {
    int gp = gp0 + tid;
    bS[tid] = gp / TT;
    cellS[0][tid] = idx[0 * NPTS + gp];
    cellS[1][tid] = idx[1 * NPTS + gp];
    cellS[2][tid] = idx[2 * NPTS + gp];
  }
  __syncthreads();

  // concat left half: copy net tile (vectorized by 8 halves)
  for (int e8 = tid; e8 < MT * 128 / 8; e8 += 256) {
    int e = e8 * 8;
    int pt = e >> 7, c = e & 127;
    *(v8h*)&xs[(size_t)pt * 256 + c] = *(const v8h*)&netIn[(size_t)(gp0 + pt) * 128 + c];
  }
  // concat right half: pooled = sum over planes of decoded scatter-max
  for (int e = tid; e < MT * 128; e += 256) {
    int pt = e >> 7, c = e & 127;
    int b = bS[pt];
    float s = 0.f;
#pragma unroll
    for (int pl = 0; pl < 3; ++pl) {
      size_t g = ((size_t)(pl * BB + b) * R2 + cellS[pl][pt]) * 128 + c;
      s += dec_f32(grid[g]);
    }
    xs[(size_t)pt * 256 + 128 + c] = (_Float16)s;
  }
  __syncthreads();

  resnet_tile(xs, ns, w0, w1, wsc, b0, b1, netOut, gp0);
}

__global__ void k_counts(const int* __restrict__ idx, int* __restrict__ cnt) {
  int gp = blockIdx.x * blockDim.x + threadIdx.x;
  if (gp >= NPTS) return;
  int b = gp / TT;
#pragma unroll
  for (int pl = 0; pl < 3; ++pl)
    atomicAdd(&cnt[(pl * BB + b) * R2 + idx[pl * NPTS + gp]], 1);
}

__global__ void k_final(const _Float16* __restrict__ netIn, const int* __restrict__ idx,
                        const _Float16* __restrict__ wc, const float* __restrict__ bc,
                        float* __restrict__ out) {
  __shared__ __align__(16) _Float16 xs[MT * 128];
  __shared__ int cellS[3][MT];
  __shared__ int bS[MT];
  const int tid  = threadIdx.x;
  const int lane = tid & 31;
  const int wv   = tid >> 5;
  const int gp0  = blockIdx.x * MT;

  if (tid < MT) {
    int gp = gp0 + tid;
    bS[tid] = gp / TT;
    cellS[0][tid] = idx[0 * NPTS + gp];
    cellS[1][tid] = idx[1 * NPTS + gp];
    cellS[2][tid] = idx[2 * NPTS + gp];
  }
  for (int e8 = tid; e8 < MT * 128 / 8; e8 += 256) {
    int e = e8 * 8;
    *(v8h*)&xs[e] = *(const v8h*)&netIn[(size_t)gp0 * 128 + e];
  }
  __syncthreads();

  const int col   = lane & 15;
  const int rbase = (lane >> 4) * 8;
  for (int nt = wv; nt < 32; nt += 8) {   // 512 cols / 16
    int n0 = nt * 16;
    v16h bf[4];
#pragma unroll
    for (int k = 0; k < 4; ++k) bf[k] = load_b(wc, 512, 32 * k, n0, lane);
    float bias = bc[n0 + col];
#pragma unroll
    for (int ms = 0; ms < 4; ++ms) {
      v8f acc = {};
#pragma unroll
      for (int k = 0; k < 4; ++k)
        acc = wmma16(load_a(xs, 128, 16 * ms, 32 * k, lane), bf[k], acc);
#pragma unroll
      for (int r = 0; r < 8; ++r) {
        int row = 16 * ms + rbase + r;
        float v = acc[r] + bias;
        int b = bS[row];
#pragma unroll
        for (int pl = 0; pl < 3; ++pl) {
          size_t o = ((size_t)((pl * BB + b) * 512 + n0 + col)) * R2 + cellS[pl][row];
          atomicAdd(&out[o], v);
        }
      }
    }
  }
}

__global__ void k_divide(float* __restrict__ out, const int* __restrict__ cnt) {
  size_t t = (size_t)blockIdx.x * blockDim.x + threadIdx.x;
  const size_t total = (size_t)3 * BB * 512 * R2;
  if (t >= total) return;
  int cell = (int)(t & (R2 - 1));
  int rest = (int)(t >> 12);        // (pl*BB+b)*512 + f
  int b  = (rest >> 9) & 3;
  int pl = rest >> 11;
  int c = cnt[(pl * BB + b) * R2 + cell];
  out[t] = out[t] / (float)(c > 0 ? c : 1);
}

// ---------- host launch ----------
extern "C" void kernel_launch(void* const* d_in, const int* in_sizes, int n_in,
                              void* d_out, int out_size, void* d_ws, size_t ws_size,
                              hipStream_t stream) {
  (void)in_sizes; (void)n_in; (void)ws_size;
  const float* p    = (const float*)d_in[0];
  const float* wpos = (const float*)d_in[1];
  const float* bpos = (const float*)d_in[2];
  const float* fc0w = (const float*)d_in[3];
  const float* fc0b = (const float*)d_in[4];
  const float* fc1w = (const float*)d_in[5];
  const float* fc1b = (const float*)d_in[6];
  const float* scw  = (const float*)d_in[7];
  const float* fccw = (const float*)d_in[8];
  const float* fccb = (const float*)d_in[9];

  char* ws = (char*)d_ws;
  size_t off = 0;
  auto take = [&](size_t bytes) -> char* {
    char* r = ws + off;
    off = (off + bytes + 255) & ~(size_t)255;
    return r;
  };
  int*      idx  = (int*)take((size_t)3 * NPTS * 4);
  _Float16* net1 = (_Float16*)take((size_t)NPTS * 128 * 2);
  _Float16* net2 = (_Float16*)take((size_t)NPTS * 128 * 2);
  unsigned* grid = (unsigned*)take((size_t)3 * BB * R2 * 128 * 4);
  int*      cnt  = (int*)take((size_t)3 * BB * R2 * 4);
  _Float16* w0h  = (_Float16*)take((size_t)5 * 256 * 128 * 2);
  _Float16* w1h  = (_Float16*)take((size_t)5 * 128 * 128 * 2);
  _Float16* wsh  = (_Float16*)take((size_t)5 * 256 * 128 * 2);
  _Float16* wch  = (_Float16*)take((size_t)128 * 512 * 2);

  k_stage_w<<<640, 256, 0, stream>>>(fc0w, fc1w, scw, fccw, w0h, w1h, wsh, wch);
  k_block0<<<NBLK, 256, 0, stream>>>(p, wpos, bpos, w0h, w1h, wsh, fc0b, fc1b, idx, net1);

  _Float16* cur = net1; _Float16* nxt = net2;
  for (int k = 1; k < 5; ++k) {
    hipMemsetAsync(grid, 0, (size_t)3 * BB * R2 * 128 * 4, stream);
    k_scatter<<<(NPTS + 255) / 256, 256, 0, stream>>>(cur, idx, grid);
    k_block<<<NBLK, 256, 0, stream>>>(cur, grid, idx,
        w0h + (size_t)k * 256 * 128, w1h + (size_t)k * 128 * 128,
        wsh + (size_t)k * 256 * 128, fc0b + k * 128, fc1b + k * 128, nxt);
    _Float16* t = cur; cur = nxt; nxt = t;
  }

  hipMemsetAsync(d_out, 0, (size_t)out_size * 4, stream);
  hipMemsetAsync(cnt, 0, (size_t)3 * BB * R2 * 4, stream);
  k_counts<<<(NPTS + 255) / 256, 256, 0, stream>>>(idx, cnt);
  k_final<<<NBLK, 256, 0, stream>>>(cur, idx, wch, fccb, (float*)d_out);
  const size_t total = (size_t)3 * BB * 512 * R2;
  k_divide<<<(unsigned)((total + 255) / 256), 256, 0, stream>>>((float*)d_out, cnt);
}
